// MACELayer_27788438405221
// MI455X (gfx1250) — matrix operand order repaired and checked by reading, not compiled, not run on hardware
//
#include <hip/hip_runtime.h>
#include <hip/hip_bf16.h>

// ---------------------------------------------------------------------------
// MACE equivariant layer for MI455X (gfx1250, wave32, WMMA + TDM).
// Heavy GEMMs run on v_wmma_f32_16x16x32_f16; tile staging and the random
// sender-feature gather use the Tensor Data Mover (tensor_load_to_lds,
// TENSORcnt); edge scatter uses global_atomic_add_f32.
// ---------------------------------------------------------------------------

typedef __attribute__((ext_vector_type(16))) _Float16     v16h;
typedef __attribute__((ext_vector_type(8)))  float        v8f;
typedef __attribute__((ext_vector_type(4)))  unsigned int u32x4;
typedef __attribute__((ext_vector_type(8)))  int          i32x8;
typedef __attribute__((ext_vector_type(4)))  int          i32x4;

#define NNODES 50000
#define NEDGES 400000

__device__ __forceinline__ v8f vzero8() {
  v8f z;
#pragma unroll
  for (int i = 0; i < 8; ++i) z[i] = 0.0f;
  return z;
}

__device__ __forceinline__ v8f wmma16(v16h a, v16h b, v8f c) {
  return __builtin_amdgcn_wmma_f32_16x16x32_f16(
      false, a, false, b, (short)0, c, false, false);
}

// ---------------- TDM helpers (CDNA5 ISA) -----
__device__ __forceinline__ void tdm_issue(u32x4 g0, i32x8 g1, i32x4 g2, i32x4 g3) {
#if __clang_major__ >= 23
  i32x8 zz = {0, 0, 0, 0, 0, 0, 0, 0};
  __builtin_amdgcn_tensor_load_to_lds(g0, g1, g2, g3, zz, 0);
#else
  __builtin_amdgcn_tensor_load_to_lds(g0, g1, g2, g3, 0);
#endif
}

// 1-D contiguous f32 tile -> LDS (nElems <= 4096).
__device__ __forceinline__ void tdm_load_1d(unsigned ldsAddr, const void* gptr,
                                            unsigned nElems) {
  unsigned long long ga = (unsigned long long)gptr;
  u32x4 g0;
  g0[0] = 1u;                                       // count=1, no gather
  g0[1] = ldsAddr;                                  // lds_addr (bytes)
  g0[2] = (unsigned)(ga & 0xFFFFFFFFu);             // global_addr[31:0]
  g0[3] = (unsigned)((ga >> 32) & 0x01FFFFFFu) | (2u << 30);  // addr[56:32]|type=2
  i32x8 g1;
  g1[0] = (int)(2u << 16);                          // data_size = 4B
  g1[1] = (int)((nElems & 0xFFFFu) << 16);          // tensor_dim0[15:0]
  g1[2] = (int)((nElems >> 16) | (1u << 16));       // tensor_dim0[31:16], dim1=1
  g1[3] = (int)((nElems & 0xFFFFu) << 16);          // tile_dim0
  g1[4] = 0;                                        // tile_dim1=0 (1-D), tile_dim2=0
  g1[5] = (int)nElems;                              // tensor_dim0_stride lo
  g1[6] = 0;
  g1[7] = 0;
  i32x4 z = {0, 0, 0, 0};
  tdm_issue(g0, g1, z, z);
}

// Gather 8 rows of 256 f32 each: row i at base + idx[i]*256 floats -> LDS.
__device__ __forceinline__ void tdm_gather8(unsigned ldsAddr, const void* base,
                                            int i0, int i1, int i2, int i3,
                                            int i4, int i5, int i6, int i7) {
  unsigned long long ga = (unsigned long long)base;
  u32x4 g0;
  g0[0] = 1u | (1u << 30) | (1u << 31);             // count=1, 32b idx, gather
  g0[1] = ldsAddr;
  g0[2] = (unsigned)(ga & 0xFFFFFFFFu);
  g0[3] = (unsigned)((ga >> 32) & 0x01FFFFFFu) | (2u << 30);
  i32x8 g1;
  g1[0] = (int)(2u << 16);                          // data_size = 4B
  g1[1] = (int)(256u << 16);                        // tensor_dim0 = 256
  g1[2] = (int)((unsigned)(NNODES & 0xFFFF) << 16); // tensor_dim1 lo
  g1[3] = (int)((NNODES >> 16) | (256u << 16));     // tensor_dim1 hi, tile_dim0=256
  g1[4] = 8;                                        // tile_dim1 = #indices
  g1[5] = 256;                                      // tensor_dim0_stride
  g1[6] = 0;
  g1[7] = 0;
  i32x4 g2 = {i0, i1, i2, i3};
  i32x4 g3 = {i4, i5, i6, i7};
  tdm_issue(g0, g1, g2, g3);
}

__device__ __forceinline__ unsigned lds_off(const void* p) {
  return (unsigned)(unsigned long long)(size_t)p;
}

// ---------------- WMMA fragment helpers ------------------------------------
// A fragment (16-bit A 16x32 layout), generic strides:
// element(row,k) = X[row*rowStride + k*colStride] * scale.
__device__ __forceinline__ v16h load_A_frag_s(const float* X, int rowStride,
                                              int colStride, int kstep, int lane,
                                              float scale) {
  const int row  = lane & 15;
  const int koff = ((lane & 16) ? 8 : 0) + kstep * 32;
  v16h a;
#pragma unroll
  for (int j = 0; j < 16; ++j) {
    int k = koff + j + ((j & 8) ? 8 : 0);
    a[j] = (_Float16)(X[row * rowStride + k * colStride] * scale);
  }
  return a;
}

// Self-connection A: A[n, k*64+c] = w[n,k] * x[n,c].
__device__ __forceinline__ v16h load_A_sc(const float* X, const float* Wsp,
                                          int kstep, int lane) {
  const int row  = lane & 15;
  const int koff = ((lane & 16) ? 8 : 0) + kstep * 32;
  v16h a;
#pragma unroll
  for (int j = 0; j < 16; ++j) {
    int kk = koff + j + ((j & 8) ? 8 : 0);
    a[j] = (_Float16)(Wsp[row * 16 + (kk >> 6)] * X[row * 64 + (kk & 63)]);
  }
  return a;
}

__device__ __forceinline__ v16h load_B_frag(const _Float16* pB, int NT,
                                            int kstep, int nt, int lane) {
  return *(const v16h*)(pB + (((size_t)(kstep * NT + nt) * 32 + lane) * 16));
}

__device__ __forceinline__ void store_D(float* Y, int ldc, int nt, int lane, v8f acc) {
  const int col   = nt * 16 + (lane & 15);
  const int rbase = (lane & 16) ? 8 : 0;
#pragma unroll
  for (int f = 0; f < 8; ++f) Y[(rbase + f) * ldc + col] = acc[f];
}

// ---------------- workspace layout ----------------
#define ACC_ELEMS     (NNODES * 256)
#define SV_OFF        ACC_ELEMS
#define PACK_BYTE_OFF ((size_t)(2 * NNODES * 256) * 4)
#define P_W0IN   0
#define P_W1IN   4096
#define P_W0OUT  8192
#define P_W1OUT  12288
#define P_WRES0  16384
#define P_WRES1  20480
#define P_WR2    24576
#define P_WSC0   40960
#define P_WSC1   106496
#define PACK_TOTAL 172032

__device__ __forceinline__ void pack_one(_Float16* dst, const float* src,
                                         int Ncols, int t) {
  const int j    = t & 15;
  const int lane = (t >> 4) & 31;
  const int NT   = Ncols >> 4;
  const int rest = t >> 9;
  const int nt   = rest % NT;
  const int ks   = rest / NT;
  const int srcK = ks * 32 + ((lane & 16) ? 8 : 0) + j + ((j & 8) ? 8 : 0);
  const int srcN = nt * 16 + (lane & 15);
  dst[t] = (_Float16)src[srcK * Ncols + srcN];
}

__global__ void prep_kernel(const float* W0_in, const float* W1_in,
                            const float* W0_out, const float* W1_out,
                            const float* Wres0, const float* Wres1,
                            const float* Wr2, const float* Wsc0, const float* Wsc1,
                            float* acc, _Float16* packed) {
  const long long total = (long long)ACC_ELEMS + PACK_TOTAL;
  for (long long i = blockIdx.x * (long long)blockDim.x + threadIdx.x;
       i < total; i += (long long)gridDim.x * blockDim.x) {
    if (i < ACC_ELEMS) { acc[i] = 0.0f; continue; }
    int p = (int)(i - ACC_ELEMS);
    if      (p < P_W1IN)   pack_one(packed + P_W0IN,  W0_in,  64,  p - P_W0IN);
    else if (p < P_W0OUT)  pack_one(packed + P_W1IN,  W1_in,  64,  p - P_W1IN);
    else if (p < P_W1OUT)  pack_one(packed + P_W0OUT, W0_out, 64,  p - P_W0OUT);
    else if (p < P_WRES0)  pack_one(packed + P_W1OUT, W1_out, 64,  p - P_W1OUT);
    else if (p < P_WRES1)  pack_one(packed + P_WRES0, Wres0,  64,  p - P_WRES0);
    else if (p < P_WR2)    pack_one(packed + P_WRES1, Wres1,  64,  p - P_WRES1);
    else if (p < P_WSC0)   pack_one(packed + P_WR2,   Wr2,    256, p - P_WR2);
    else if (p < P_WSC1)   pack_one(packed + P_WSC0,  Wsc0,   64,  p - P_WSC0);
    else                   pack_one(packed + P_WSC1,  Wsc1,   64,  p - P_WSC1);
  }
}

// Kernel 1: input linears. TDM stages the node tiles; A-frags read from LDS.
__global__ void __launch_bounds__(32)
node_in_kernel(const float* ns, const float* nv, const _Float16* packed, float* sv) {
  __shared__ __align__(16) float NST[16 * 64];
  __shared__ __align__(16) float NVT[16 * 192];
  const int node0 = blockIdx.x * 16;
  const int lane  = threadIdx.x;

  tdm_load_1d(lds_off(NST), ns + (size_t)node0 * 64, 1024);
  tdm_load_1d(lds_off(NVT), nv + (size_t)node0 * 192, 3072);
  __builtin_amdgcn_s_wait_tensorcnt(0);
  __syncthreads();

  for (int comp = 0; comp < 4; ++comp) {
    const float* Xp = (comp == 0) ? NST : (NVT + (comp - 1));
    const int rs = (comp == 0) ? 64 : 192;
    const int cs = (comp == 0) ? 1 : 3;
    const _Float16* pB = packed + (comp == 0 ? P_W0IN : P_W1IN);
    v8f a4[4] = {vzero8(), vzero8(), vzero8(), vzero8()};
#pragma unroll
    for (int s = 0; s < 2; ++s) {
      v16h a = load_A_frag_s(Xp, rs, cs, s, lane, 1.0f);
#pragma unroll
      for (int nt = 0; nt < 4; ++nt)
        a4[nt] = wmma16(a, load_B_frag(pB, 4, s, nt, lane), a4[nt]);
    }
#pragma unroll
    for (int nt = 0; nt < 4; ++nt) {
      int col   = nt * 16 + (lane & 15);
      int rbase = (lane & 16) ? 8 : 0;
#pragma unroll
      for (int f = 0; f < 8; ++f)
        sv[(size_t)(node0 + rbase + f) * 256 + comp * 64 + col] = a4[nt][f];
    }
  }
}

// Kernel 2: TDM-gather sender rows, radial MLP (layer2 WMMA), scatter atomics.
__global__ void __launch_bounds__(32)
edge_kernel(const float* vectors, const float* rembed, const float* Wr1,
            const float* br1, const _Float16* packed, const float* sv,
            const int* senders, const int* receivers, float* acc) {
  __shared__ __align__(16) float SVT[16 * 256];   // gathered sender features
  __shared__ float hid[16 * 64];
  __shared__ float rw[16 * 256];
  __shared__ float uu[16][3];
  const int e0   = blockIdx.x * 16;
  const int lane = threadIdx.x;

  // Uniform sender indices -> two TDM gather ops (overlap with MLP below).
  int sidx[16];
#pragma unroll
  for (int i = 0; i < 16; ++i) sidx[i] = senders[e0 + i];
  tdm_gather8(lds_off(SVT), sv, sidx[0], sidx[1], sidx[2], sidx[3],
              sidx[4], sidx[5], sidx[6], sidx[7]);
  tdm_gather8(lds_off(SVT) + 8 * 256 * 4, sv, sidx[8], sidx[9], sidx[10], sidx[11],
              sidx[12], sidx[13], sidx[14], sidx[15]);

  if (lane < 16) {
    int e = e0 + lane;
    float x = vectors[e * 3 + 0], y = vectors[e * 3 + 1], z = vectors[e * 3 + 2];
    float inv = 1.0f / (sqrtf(x * x + y * y + z * z) + 1e-8f);
    uu[lane][0] = x * inv; uu[lane][1] = y * inv; uu[lane][2] = z * inv;
  }
  // hidden = silu(re @ Wr1 + br1)   (K=8: VALU)
  for (int t = lane; t < 1024; t += 32) {
    int e = t >> 6, h = t & 63;
    float a = br1[h];
#pragma unroll
    for (int r = 0; r < 8; ++r) a += rembed[(e0 + e) * 8 + r] * Wr1[r * 64 + h];
    hid[t] = a / (1.0f + expf(-a));
  }
  __syncthreads();
  // rw = hidden @ Wr2 : [16,64]x[64,256]
  const _Float16* pB = packed + P_WR2;
  for (int g = 0; g < 4; ++g) {
    v8f a4[4] = {vzero8(), vzero8(), vzero8(), vzero8()};
#pragma unroll
    for (int s = 0; s < 2; ++s) {
      v16h a = load_A_frag_s(hid, 64, 1, s, lane, 1.0f);
#pragma unroll
      for (int q = 0; q < 4; ++q)
        a4[q] = wmma16(a, load_B_frag(pB, 16, s, g * 4 + q, lane), a4[q]);
    }
#pragma unroll
    for (int q = 0; q < 4; ++q) store_D(rw, 256, g * 4 + q, lane, a4[q]);
  }
  __builtin_amdgcn_s_wait_tensorcnt(0);   // sender rows landed in SVT
  __syncthreads();
  // messages + scatter (rw row = [p0|p1|p2|p3], SVT row = [s|v0|v1|v2])
  for (int e = 0; e < 16; ++e) {
    const int   rI = receivers[e0 + e];
    const float u0 = uu[e][0], u1 = uu[e][1], u2 = uu[e][2];
#pragma unroll
    for (int cc = 0; cc < 2; ++cc) {
      const int c  = lane + cc * 32;
      const float se = SVT[e * 256 + c];
      const float v0 = SVT[e * 256 + 64 + c];
      const float v1 = SVT[e * 256 + 128 + c];
      const float v2 = SVT[e * 256 + 192 + c];
      const float r0 = rw[e * 256 + c],       r1 = rw[e * 256 + 64 + c];
      const float r2 = rw[e * 256 + 128 + c], r3 = rw[e * 256 + 192 + c];
      const float dv = v0 * u0 + v1 * u1 + v2 * u2;
      unsafeAtomicAdd(&acc[(size_t)rI * 256 + c],       r0 * se + r1 * dv);
      unsafeAtomicAdd(&acc[(size_t)rI * 256 + 64 + c],  r2 * se * u0 + r3 * v0);
      unsafeAtomicAdd(&acc[(size_t)rI * 256 + 128 + c], r2 * se * u1 + r3 * v1);
      unsafeAtomicAdd(&acc[(size_t)rI * 256 + 192 + c], r2 * se * u2 + r3 * v2);
    }
  }
}

// Kernel 3: output linear -> self-connection -> E3LayerNorm -> residual.
__global__ void __launch_bounds__(32)
node_out_kernel(const float* acc, const float* ns, const float* nv,
                const float* species_embed, const int* node_species,
                const float* g0, const float* g1,
                const _Float16* packed, float* out) {
  __shared__ __align__(16) float X[4 * 16 * 64];
  __shared__ __align__(16) float S1[16 * 256];    // ACC tile, later RES
  __shared__ __align__(16) float NST[16 * 64];
  __shared__ __align__(16) float NVT[16 * 192];
  __shared__ float Wsp[16 * 16];
  const int node0 = blockIdx.x * 16;
  const int lane  = threadIdx.x;
  const float INV_SQRT_DEG = 0.35355339059327373f;

  tdm_load_1d(lds_off(S1),  acc + (size_t)node0 * 256, 4096);
  tdm_load_1d(lds_off(NST), ns + (size_t)node0 * 64,   1024);
  tdm_load_1d(lds_off(NVT), nv + (size_t)node0 * 192,  3072);

  for (int t = lane; t < 256; t += 32) {
    int r = t >> 4, k = t & 15;
    Wsp[t] = species_embed[node_species[node0 + r] * 16 + k];
  }
  __builtin_amdgcn_s_wait_tensorcnt(0);
  __syncthreads();

  // phase 1: x = (acc * inv_sqrt_deg) @ Wout   (A direct from S1)
  for (int comp = 0; comp < 4; ++comp) {
    const _Float16* pB = packed + (comp == 0 ? P_W0OUT : P_W1OUT);
    v8f a4[4] = {vzero8(), vzero8(), vzero8(), vzero8()};
#pragma unroll
    for (int s = 0; s < 2; ++s) {
      v16h a = load_A_frag_s(S1 + comp * 64, 256, 1, s, lane, INV_SQRT_DEG);
#pragma unroll
      for (int nt = 0; nt < 4; ++nt)
        a4[nt] = wmma16(a, load_B_frag(pB, 4, s, nt, lane), a4[nt]);
    }
#pragma unroll
    for (int nt = 0; nt < 4; ++nt) store_D(X + comp * 1024, 64, nt, lane, a4[nt]);
  }
  __syncthreads();
  // phase 2: species self-connection, x <- (x (x) w) @ Wsc  (K = 1024)
  for (int comp = 0; comp < 4; ++comp) {
    const _Float16* pB = packed + (comp == 0 ? P_WSC0 : P_WSC1);
    v8f a4[4] = {vzero8(), vzero8(), vzero8(), vzero8()};
    for (int s = 0; s < 32; ++s) {
      v16h a = load_A_sc(X + comp * 1024, Wsp, s, lane);
#pragma unroll
      for (int nt = 0; nt < 4; ++nt)
        a4[nt] = wmma16(a, load_B_frag(pB, 4, s, nt, lane), a4[nt]);
    }
    __syncthreads();
#pragma unroll
    for (int nt = 0; nt < 4; ++nt) store_D(X + comp * 1024, 64, nt, lane, a4[nt]);
    __syncthreads();
  }
  // phase 3: residual adapter GEMMs (A direct from NST/NVT, D -> S1 as RES)
  for (int comp = 0; comp < 4; ++comp) {
    const float* Xp = (comp == 0) ? NST : (NVT + (comp - 1));
    const int rs = (comp == 0) ? 64 : 192;
    const int cs = (comp == 0) ? 1 : 3;
    const _Float16* pB = packed + (comp == 0 ? P_WRES0 : P_WRES1);
    v8f a4[4] = {vzero8(), vzero8(), vzero8(), vzero8()};
#pragma unroll
    for (int s = 0; s < 2; ++s) {
      v16h a = load_A_frag_s(Xp, rs, cs, s, lane, 1.0f);
#pragma unroll
      for (int nt = 0; nt < 4; ++nt)
        a4[nt] = wmma16(a, load_B_frag(pB, 4, s, nt, lane), a4[nt]);
    }
#pragma unroll
    for (int nt = 0; nt < 4; ++nt) store_D(S1 + comp * 1024, 64, nt, lane, a4[nt]);
  }
  __syncthreads();
  // phase 4: E3LayerNorm + residual + store (lane r owns node node0+r)
  if (lane < 16) {
    const int n = node0 + lane;
    const float* x0 = X + lane * 64;
    float mu = 0.0f, m2 = 0.0f;
#pragma unroll
    for (int c = 0; c < 64; ++c) { float t = x0[c]; mu += t; m2 += t * t; }
    mu *= (1.0f / 64.0f);
    float var  = m2 * (1.0f / 64.0f) - mu * mu;
    float rstd = rsqrtf(var + 1e-6f);
    float ss = 0.0f;
#pragma unroll
    for (int c = 0; c < 64; ++c) {
      float a0 = X[1024 + lane * 64 + c];
      float a1 = X[2048 + lane * 64 + c];
      float a2 = X[3072 + lane * 64 + c];
      ss += a0 * a0 + a1 * a1 + a2 * a2;
    }
    float rinv = rsqrtf(ss * (1.0f / 64.0f) + 1e-6f);
#pragma unroll
    for (int c = 0; c < 64; ++c)
      out[(size_t)n * 256 + c] = (x0[c] - mu) * rstd * g0[c] + S1[lane * 64 + c];
#pragma unroll
    for (int c = 0; c < 64; ++c) {
      float gs = rinv * g1[c];
#pragma unroll
      for (int i = 0; i < 3; ++i)
        out[(size_t)n * 256 + 64 + c * 3 + i] =
            X[(1 + i) * 1024 + lane * 64 + c] * gs + S1[(1 + i) * 1024 + lane * 64 + c];
    }
  }
}

extern "C" void kernel_launch(void* const* d_in, const int* in_sizes, int n_in,
                              void* d_out, int out_size, void* d_ws, size_t ws_size,
                              hipStream_t stream) {
  (void)in_sizes; (void)n_in; (void)out_size; (void)ws_size;
  const float* vectors   = (const float*)d_in[0];
  const float* ns        = (const float*)d_in[1];
  const float* nv        = (const float*)d_in[2];
  const float* rembed    = (const float*)d_in[3];
  const float* spemb     = (const float*)d_in[4];
  const float* W0_in     = (const float*)d_in[5];
  const float* W1_in     = (const float*)d_in[6];
  const float* Wr1       = (const float*)d_in[7];
  const float* br1       = (const float*)d_in[8];
  const float* Wr2       = (const float*)d_in[9];
  const float* W0_out    = (const float*)d_in[10];
  const float* W1_out    = (const float*)d_in[11];
  const float* Wsc0      = (const float*)d_in[12];
  const float* Wsc1      = (const float*)d_in[13];
  const float* g0        = (const float*)d_in[14];
  const float* g1        = (const float*)d_in[15];
  const float* Wres0     = (const float*)d_in[16];
  const float* Wres1     = (const float*)d_in[17];
  const int*   nspecies  = (const int*)d_in[18];
  const int*   receivers = (const int*)d_in[19];
  const int*   senders   = (const int*)d_in[20];
  float* out = (float*)d_out;

  float*    acc    = (float*)d_ws;
  float*    sv     = (float*)d_ws + SV_OFF;
  _Float16* packed = (_Float16*)((char*)d_ws + PACK_BYTE_OFF);

  prep_kernel<<<2048, 256, 0, stream>>>(W0_in, W1_in, W0_out, W1_out,
                                        Wres0, Wres1, Wr2, Wsc0, Wsc1,
                                        acc, packed);
  node_in_kernel<<<NNODES / 16, 32, 0, stream>>>(ns, nv, packed, sv);
  edge_kernel<<<NEDGES / 16, 32, 0, stream>>>(vectors, rembed, Wr1, br1, packed,
                                              sv, senders, receivers, acc);
  node_out_kernel<<<NNODES / 16, 32, 0, stream>>>(acc, ns, nv, spemb, nspecies,
                                                  g0, g1, packed, out);
}
